// MHCrossAttention_77309411787
// MI455X (gfx1250) — matrix operand order repaired and checked by reading, not compiled, not run on hardware
//
#include <hip/hip_runtime.h>
#include <hip/hip_bf16.h>

// ---------------------------------------------------------------------------
// MH cross-attention, CDNA5 (gfx1250, wave32) WMMA implementation.
// B=4, T=T_CTX=2048, E=512, H=8, s=64.
// ---------------------------------------------------------------------------

typedef __bf16 v8bf  __attribute__((ext_vector_type(8)));
typedef __bf16 v16bf __attribute__((ext_vector_type(16)));
typedef float  v8f   __attribute__((ext_vector_type(8)));

#define NB   4
#define NT   2048
#define NE   512
#define NH   8
#define NS   64
#define NROWS (NB * NT)          // 8192

static __device__ inline v8f wmma_bf16(v16bf a, v16bf b, v8f c) {
  return __builtin_amdgcn_wmma_f32_16x16x32_bf16(
      /*neg_a=*/false, a, /*neg_b=*/false, b,
      /*c_mod=*/(short)0, c, /*reuse_a=*/false, /*reuse_b=*/false);
}

// Combine the two 8-element halves of an A-matrix lane fragment.
static __device__ inline v16bf ld_a16(const __bf16* lo, const __bf16* hi) {
  v8bf a = *(const v8bf*)lo;
  v8bf b = *(const v8bf*)hi;
  return __builtin_shufflevector(a, b, 0, 1, 2, 3, 4, 5, 6, 7,
                                 8, 9, 10, 11, 12, 13, 14, 15);
}

// ---------------------------------------------------------------------------
// DPP row_ror butterfly reductions within each 16-lane row (VALU only, no LDS).
// row_ror:N dpp_ctrl encoding = 0x120 | N.
// ---------------------------------------------------------------------------
template <int CTRL>
static __device__ inline float dpp_rot(float v) {
  int i = __float_as_int(v);
  int r = __builtin_amdgcn_update_dpp(i, i, CTRL, 0xf, 0xf, false);
  return __int_as_float(r);
}
static __device__ inline float red_max16(float x) {
  x = fmaxf(x, dpp_rot<0x128>(x));
  x = fmaxf(x, dpp_rot<0x124>(x));
  x = fmaxf(x, dpp_rot<0x122>(x));
  x = fmaxf(x, dpp_rot<0x121>(x));
  return x;
}
static __device__ inline float red_sum16(float x) {
  x += dpp_rot<0x128>(x);
  x += dpp_rot<0x124>(x);
  x += dpp_rot<0x122>(x);
  x += dpp_rot<0x121>(x);
  return x;
}

// ---------------------------------------------------------------------------
// f32 -> bf16 conversion with scale (bakes 512^-0.25 [and log2e] into Wq/Wk)
// ---------------------------------------------------------------------------
__global__ __launch_bounds__(256) void cvtk(const float* __restrict__ s,
                                            __bf16* __restrict__ d,
                                            int n, float scale) {
  int i = blockIdx.x * 256 + threadIdx.x;
  if (i < n) d[i] = (__bf16)(s[i] * scale);
}

// ---------------------------------------------------------------------------
// GEMM: out[n][m] = sum_k A[n][k] * W[m][k]   (A: NROWS x 512, W: 512 x 512, bf16)
// Each wave: one 32-row x 64-col tile (2 row-tiles share every B fragment).
// Block = 8 waves = all 512 cols of 32 rows.  Grid = NROWS/32.
// MODE 0: store bf16 head-scattered   out[b][h][t][s]   (Q / K)
// MODE 1: store bf16 transposed       out[b][h][s][c]   (V^T)
// MODE 2: store f32 + bias            out[n][m]         (final U projection)
// ---------------------------------------------------------------------------
template <int MODE>
__global__ __launch_bounds__(256) void gemm_wmma(const __bf16* __restrict__ A,
                                                 const __bf16* __restrict__ W,
                                                 void* __restrict__ outp,
                                                 const float* __restrict__ bias) {
  const int lane  = threadIdx.x & 31;
  const int wv    = threadIdx.x >> 5;     // 0..7 -> 64-col group
  const int lrow  = lane & 15;
  const int lhalf = lane >> 4;
  const int row0  = blockIdx.x * 32;
  const int m0    = wv * 64;
  const int akb   = lhalf * 16;           // A K-offset for this half-wave

  v8f acc[2][4];
#pragma unroll
  for (int r = 0; r < 2; ++r)
#pragma unroll
    for (int i = 0; i < 4; ++i) acc[r][i] = v8f{};

  for (int k0 = 0; k0 < NE; k0 += 32) {
    v16bf a[2];
#pragma unroll
    for (int r = 0; r < 2; ++r) {
      const __bf16* arow = A + (size_t)(row0 + r * 16 + lrow) * NE + k0;
      a[r] = ld_a16(arow + akb, arow + 8 + akb);
    }
    // B fragments: col m = m0 + i*16 + lrow, K = k0 + lhalf*16 .. +15
    const __bf16* wp = W + (size_t)(m0 + lrow) * NE + k0 + lhalf * 16;
#pragma unroll
    for (int i = 0; i < 4; ++i) {
      v16bf b = *(const v16bf*)(wp + (size_t)i * 16 * NE);
#pragma unroll
      for (int r = 0; r < 2; ++r) acc[r][i] = wmma_bf16(a[r], b, acc[r][i]);
    }
  }

#pragma unroll
  for (int r = 0; r < 2; ++r) {
    const int rbase = row0 + r * 16 + lhalf * 8; // first of this lane's 8 C rows
    const int b_ = rbase >> 11;                  // row / 2048
#pragma unroll
    for (int i = 0; i < 4; ++i) {
      const int m = m0 + i * 16 + lrow;          // C layout: this lane's column
      if (MODE == 0) {
        const int h = m >> 6, s = m & 63;
        __bf16* q = (__bf16*)outp + (((size_t)b_ * NH + h) * NT) * NS + s;
#pragma unroll
        for (int j = 0; j < 8; ++j) {
          const int t = (rbase + j) & (NT - 1);
          q[(size_t)t * NS] = (__bf16)acc[r][i][j];
        }
      } else if (MODE == 1) {
        const int h = m >> 6, s = m & 63;
        const int c = rbase & (NT - 1);
        __bf16* vt = (__bf16*)outp + (((size_t)b_ * NH + h) * NS + s) * NT + c;
        v8bf pv;
#pragma unroll
        for (int j = 0; j < 8; ++j) pv[j] = (__bf16)acc[r][i][j];
        *(v8bf*)vt = pv;                         // 8 consecutive c values
      } else {
        float* op = (float*)outp;
        const float bv = bias[m];
#pragma unroll
        for (int j = 0; j < 8; ++j)
          op[(size_t)(rbase + j) * NE + m] = acc[r][i][j] + bv;
      }
    }
  }
}

// ---------------------------------------------------------------------------
// Flash attention. One wave = 32 query rows (2 row-tiles) of one (b,h); both
// row-tiles share every K/V fragment (2x arithmetic intensity). Context is
// streamed in chunks of 32: 8 score WMMAs, exp2-domain online softmax with
// DPP-row max reduction and DEFERRED l-sum (per-lane partials, one cross-lane
// reduction at the end), P -> LDS -> A-layout, 8 P*V WMMAs.
// Qb: bf16 [bh][t][64] pre-scaled by 512^-0.25 * log2(e); Kb scaled 512^-0.25.
// Vt: bf16 [bh][64][c].  Ob: bf16 [b][t][512] (head-merged).
// ---------------------------------------------------------------------------
__global__ __launch_bounds__(256) void attn_kernel(const __bf16* __restrict__ Qb,
                                                   const __bf16* __restrict__ Kb,
                                                   const __bf16* __restrict__ Vt,
                                                   const int* __restrict__ mask,
                                                   __bf16* __restrict__ Ob) {
  __shared__ __attribute__((aligned(32))) __bf16 plds[8][2][16 * 32];

  const int lane  = threadIdx.x & 31;
  const int wv    = threadIdx.x >> 5;
  const int w     = blockIdx.x * 8 + wv;  // global wave tile
  const int bh    = w >> 6;               // 0..31  (64 x 32-row tiles per bh)
  const int rt    = w & 63;               // 32-row query tile
  const int lrow  = lane & 15;
  const int lhalf = lane >> 4;
  const int akb   = lhalf * 16;
  const int row0  = rt * 32;

  // Q A-fragments: 2 row-tiles x 2 K-tiles, loaded once.
  v16bf qa[2][2];
#pragma unroll
  for (int r = 0; r < 2; ++r) {
    const __bf16* qrow = Qb + ((size_t)bh * NT + row0 + r * 16 + lrow) * NS;
    qa[r][0] = ld_a16(qrow + akb,      qrow + 8 + akb);
    qa[r][1] = ld_a16(qrow + 32 + akb, qrow + 40 + akb);
  }

  // Row mask bias (broadcast over context axis).
  float rbias[2][8];
#pragma unroll
  for (int r = 0; r < 2; ++r) {
    const int rb = row0 + r * 16 + lhalf * 8;
#pragma unroll
    for (int j = 0; j < 8; ++j)
      rbias[r][j] = mask[bh * NT + rb + j] ? 0.f : -__builtin_inff();
  }

  v8f o[2][4];
  float mrow[2][8], lp[2][8];
#pragma unroll
  for (int r = 0; r < 2; ++r) {
#pragma unroll
    for (int i = 0; i < 4; ++i) o[r][i] = v8f{};
#pragma unroll
    for (int j = 0; j < 8; ++j) { mrow[r][j] = -__builtin_inff(); lp[r][j] = 0.f; }
  }

  const __bf16* kb = Kb + (size_t)bh * NT * NS;
  const __bf16* vb = Vt + (size_t)bh * NS * NT;

  for (int c0 = 0; c0 < NT; c0 += 32) {
    // --- K fragments for this chunk (shared by both row-tiles)
    const __bf16* kr0 = kb + (size_t)(c0 + lrow) * NS + lhalf * 16;
    v16bf kf0 = *(const v16bf*)kr0;
    v16bf kf1 = *(const v16bf*)(kr0 + 32);
    v16bf kf2 = *(const v16bf*)(kr0 + 16 * NS);
    v16bf kf3 = *(const v16bf*)(kr0 + 16 * NS + 32);

    // --- scores (already in log2 domain): S = Q * K^T
    v8f s[2][2];
#pragma unroll
    for (int r = 0; r < 2; ++r) {
      v8f t0 = v8f{}, t1 = v8f{};
      t0 = wmma_bf16(qa[r][0], kf0, t0);
      t0 = wmma_bf16(qa[r][1], kf1, t0);
      t1 = wmma_bf16(qa[r][0], kf2, t1);
      t1 = wmma_bf16(qa[r][1], kf3, t1);
      s[r][0] = t0; s[r][1] = t1;
    }

    // --- online softmax: DPP row-max, deferred per-lane partial sums
#pragma unroll
    for (int r = 0; r < 2; ++r) {
      __bf16* pw = &plds[wv][r][0];
#pragma unroll
      for (int j = 0; j < 8; ++j) {
        const float a0 = s[r][0][j] + rbias[r][j];
        const float a1 = s[r][1][j] + rbias[r][j];
        const float t  = red_max16(fmaxf(a0, a1));
        const float mn = fmaxf(mrow[r][j], t);
        const float al = __builtin_amdgcn_exp2f(mrow[r][j] - mn);
        mrow[r][j] = mn;
        const float p0 = __builtin_amdgcn_exp2f(a0 - mn);
        const float p1 = __builtin_amdgcn_exp2f(a1 - mn);
        lp[r][j] = lp[r][j] * al + (p0 + p1);   // per-lane partial row sum
        o[r][0][j] *= al; o[r][1][j] *= al; o[r][2][j] *= al; o[r][3][j] *= al;
        pw[(lhalf * 8 + j) * 32 + lrow]      = (__bf16)p0;
        pw[(lhalf * 8 + j) * 32 + 16 + lrow] = (__bf16)p1;
      }
    }
    // Same-wave LDS RAW: DS ops are in-order per wave; wait to be safe.
    asm volatile("s_wait_dscnt 0" ::: "memory");

    // --- V fragments (shared by both row-tiles), P*V accumulation
    const __bf16* vr = vb + (size_t)lrow * NT + c0 + lhalf * 16;
    v16bf vf0 = *(const v16bf*)(vr + (size_t)0 * 16 * NT);
    v16bf vf1 = *(const v16bf*)(vr + (size_t)1 * 16 * NT);
    v16bf vf2 = *(const v16bf*)(vr + (size_t)2 * 16 * NT);
    v16bf vf3 = *(const v16bf*)(vr + (size_t)3 * 16 * NT);
#pragma unroll
    for (int r = 0; r < 2; ++r) {
      const __bf16* pr = &plds[wv][r][0] + lrow * 32;
      v16bf pa = ld_a16(pr + akb, pr + 8 + akb);
      o[r][0] = wmma_bf16(pa, vf0, o[r][0]);
      o[r][1] = wmma_bf16(pa, vf1, o[r][1]);
      o[r][2] = wmma_bf16(pa, vf2, o[r][2]);
      o[r][3] = wmma_bf16(pa, vf3, o[r][3]);
    }
  }

  // --- one cross-lane sum reduction, normalize, store [b][t][h*64+s]
  const int b_ = bh >> 3, h = bh & 7;
#pragma unroll
  for (int r = 0; r < 2; ++r) {
    float inv_[8];
#pragma unroll
    for (int j = 0; j < 8; ++j) inv_[j] = 1.f / red_sum16(lp[r][j]);
#pragma unroll
    for (int i = 0; i < 4; ++i) {
      const int s_ = i * 16 + lrow;
#pragma unroll
      for (int j = 0; j < 8; ++j) {
        const int t = row0 + r * 16 + lhalf * 8 + j;
        Ob[((size_t)b_ * NT + t) * NE + h * NS + s_] =
            (__bf16)(o[r][i][j] * inv_[j]);
      }
    }
  }
}

// ---------------------------------------------------------------------------
// Host-side launch
// ---------------------------------------------------------------------------
extern "C" void kernel_launch(void* const* d_in, const int* in_sizes, int n_in,
                              void* d_out, int out_size, void* d_ws, size_t ws_size,
                              hipStream_t stream) {
  const float* x    = (const float*)d_in[0];
  const float* ctx  = (const float*)d_in[1];
  const int*   mask = (const int*)d_in[2];
  const float* Wq   = (const float*)d_in[3];
  const float* Wk   = (const float*)d_in[4];
  const float* Wv   = (const float*)d_in[5];
  const float* Wu   = (const float*)d_in[6];
  const float* bu   = (const float*)d_in[7];
  float* out = (float*)d_out;

  char* ws = (char*)d_ws;
  size_t o = 0;
  const size_t big = (size_t)NROWS * NE * sizeof(__bf16);   // 8 MiB
  const size_t wsz = (size_t)NE * NE * sizeof(__bf16);      // 512 KiB
  __bf16* Xb  = (__bf16*)(ws + o); o += big;
  __bf16* Cb  = (__bf16*)(ws + o); o += big;
  __bf16* Qb  = (__bf16*)(ws + o); o += big;
  __bf16* Kb  = (__bf16*)(ws + o); o += big;
  __bf16* Vt  = (__bf16*)(ws + o); o += big;
  __bf16* Ob  = (__bf16*)(ws + o); o += big;
  __bf16* Wqb = (__bf16*)(ws + o); o += wsz;
  __bf16* Wkb = (__bf16*)(ws + o); o += wsz;
  __bf16* Wvb = (__bf16*)(ws + o); o += wsz;
  __bf16* Wub = (__bf16*)(ws + o); o += wsz;

  const float inv4  = 0.21022410381342865f;   // 512^(-0.25)
  const float log2e = 1.4426950408889634f;    // scores computed in log2 domain

  const int nin = NROWS * NE;                 // 4,194,304
  const int nw  = NE * NE;                    // 262,144
  cvtk<<<nin / 256, 256, 0, stream>>>(x,   Xb, nin, 1.f);
  cvtk<<<nin / 256, 256, 0, stream>>>(ctx, Cb, nin, 1.f);
  cvtk<<<nw  / 256, 256, 0, stream>>>(Wq, Wqb, nw, inv4 * log2e);
  cvtk<<<nw  / 256, 256, 0, stream>>>(Wk, Wkb, nw, inv4);
  cvtk<<<nw  / 256, 256, 0, stream>>>(Wv, Wvb, nw, 1.f);
  cvtk<<<nw  / 256, 256, 0, stream>>>(Wu, Wub, nw, 1.f);

  const int gblocks = NROWS / 32;             // 256
  gemm_wmma<0><<<gblocks, 256, 0, stream>>>(Xb, Wqb, (void*)Qb, nullptr);
  gemm_wmma<0><<<gblocks, 256, 0, stream>>>(Cb, Wkb, (void*)Kb, nullptr);
  gemm_wmma<1><<<gblocks, 256, 0, stream>>>(Cb, Wvb, (void*)Vt, nullptr);

  // 32 (b,h) * 64 query tiles (32 rows each) = 2048 waves / 8 per block
  attn_kernel<<<256, 256, 0, stream>>>(Qb, Kb, Vt, mask, Ob);

  gemm_wmma<2><<<gblocks, 256, 0, stream>>>(Ob, Wub, (void*)out, bu);
}
